// MoEFFN_67800353734989
// MI455X (gfx1250) — matrix operand order repaired and checked by reading, not compiled, not run on hardware
//
#include <hip/hip_runtime.h>
#include <hip/hip_bf16.h>

// ---------------------------------------------------------------------------
// MoE FFN for MI455X (gfx1250), v3.
//
// Sparse FLOPs = T*K*2*(2*H*F) ~ 77 GFLOP; bf16 weights (75 MB after one-time
// convert/transpose) are fully L2-resident (192 MB). Matrix-pipe bound ->
// V_WMMA_F32_16X16X32_BF16 with f32 accumulate.
// v3: software-pipelined fragment loads (B one step ahead, A preloaded per
// chunk) to kill the s_wait_loadcnt 0x0 right before each WMMA, and a
// double-buffered H chunk so each F-chunk needs only ONE workgroup barrier.
// ---------------------------------------------------------------------------

#define H_DIM 768
#define F_DIM 3072
#define E_NUM 8
#define T_TOK 4096   // B*S
#define TM    32     // tokens per tile (2 x 16-row WMMA tiles)
#define FN    64     // F-chunk width
#define XLD   (H_DIM + 8)  // 776 bf16 rows -> 1552B stride: conflict-free 16B reads
#define HLD   (FN + 8)     // 72 bf16 rows  -> 144B stride:  conflict-free 16B reads

typedef __attribute__((ext_vector_type(16))) __bf16 v16bf;
typedef __attribute__((ext_vector_type(8)))  float  v8f;

union Frag {
  v16bf v;
  unsigned short us[16];
  uint4 q[2];
};

__device__ __forceinline__ unsigned short f2bf(float f) {
  unsigned int u = __float_as_uint(f);
  u += 0x7FFFu + ((u >> 16) & 1u);   // round-to-nearest-even
  return (unsigned short)(u >> 16);
}

__device__ __forceinline__ float gelu_tanh(float x) {
  float x3 = x * x * x;
  float t  = tanhf(0.7978845608028654f * (x + 0.044715f * x3));
  return 0.5f * x * (1.0f + t);
}

// A fragment (16x32 bf16) from a row-major LDS tile.
// ISA 7.12.2: lane L holds row M = L%16; elems 0..7 -> K=(L/16)*8+j,
// elems 8..15 -> K=16+(L/16)*8+(j-8). Two 16B ds reads per lane.
__device__ __forceinline__ v16bf frag_a_lds(const unsigned short* base, int ld,
                                            int m0, int k0, int lane) {
  int row = m0 + (lane & 15);
  int g   = lane >> 4;
  const unsigned short* p = base + row * ld + k0 + g * 8;
  Frag f;
  f.q[0] = *reinterpret_cast<const uint4*>(p);
  f.q[1] = *reinterpret_cast<const uint4*>(p + 16);
  return f.v;
}

// B fragment (32x16 bf16) from a PRE-TRANSPOSED bf16 global matrix BT[n][k].
// Lane L holds col N = L%16; elems 0..7 -> K=(L/16)*8+j,
// elems 8..15 -> K=16+(L/16)*8+(j-8): two global_load_b128 per lane.
__device__ __forceinline__ v16bf frag_b_glbT(const unsigned short* __restrict__ BT,
                                             int ldk, int k0, int n0, int lane) {
  const unsigned short* p = BT + (size_t)(n0 + (lane & 15)) * ldk
                               + k0 + ((lane >> 4) << 3);
  Frag f;
  f.q[0] = *reinterpret_cast<const uint4*>(p);
  f.q[1] = *reinterpret_cast<const uint4*>(p + 16);
  return f.v;
}

// ---------------------------------------------------------------------------
// One-time per launch: f32 -> bf16 convert + transpose (LDS-tiled).
// src: per-expert M x N f32 row-major.  dst: per-expert N x M bf16.
// ---------------------------------------------------------------------------
__global__ __launch_bounds__(256) void transpose_to_bf16_kernel(
    const float* __restrict__ src, unsigned short* __restrict__ dst,
    int M, int N)
{
  __shared__ unsigned short tile[32][33];
  const int e = blockIdx.z;
  const float* s    = src + (size_t)e * M * N;
  unsigned short* d = dst + (size_t)e * M * N;
  const int n0 = blockIdx.x * 32, m0 = blockIdx.y * 32;
  const int tx = threadIdx.x & 31, ty = threadIdx.x >> 5;   // 32 x 8

#pragma unroll
  for (int i = 0; i < 32; i += 8)
    tile[ty + i][tx] = f2bf(s[(size_t)(m0 + ty + i) * N + n0 + tx]);
  __syncthreads();
#pragma unroll
  for (int i = 0; i < 32; i += 8)
    d[(size_t)(n0 + ty + i) * M + m0 + tx] = tile[tx][ty + i];
}

// One-time per launch: x f32 -> bf16 (packed pair stores).
__global__ __launch_bounds__(256) void convert_x_kernel(
    const float* __restrict__ x, unsigned int* __restrict__ xbf2)
{
  int i = blockIdx.x * 256 + threadIdx.x;          // one uint = 2 bf16
  if (i >= T_TOK * H_DIM / 2) return;
  float a = x[2 * i], b = x[2 * i + 1];
  xbf2[i] = (unsigned int)f2bf(a) | ((unsigned int)f2bf(b) << 16);
}

// ---------------------------------------------------------------------------
// Router: one wave per token; top-2 + softmax; atomic append to expert lists.
// ---------------------------------------------------------------------------
__global__ __launch_bounds__(256) void moe_router_kernel(
    const float* __restrict__ x, const float* __restrict__ gate_w,
    int* __restrict__ cnt, int* __restrict__ toklist, float* __restrict__ wlist)
{
  const int lane = threadIdx.x & 31;
  const int wid  = threadIdx.x >> 5;
  const int t    = blockIdx.x * 8 + wid;
  if (t >= T_TOK) return;

  float acc[E_NUM];
#pragma unroll
  for (int e = 0; e < E_NUM; ++e) acc[e] = 0.0f;

  const float* xr = x + (size_t)t * H_DIM;
  for (int h = lane; h < H_DIM; h += 32) {
    float xv = xr[h];
    const float* g = gate_w + h * E_NUM;
#pragma unroll
    for (int e = 0; e < E_NUM; ++e) acc[e] += xv * g[e];
  }
#pragma unroll
  for (int off = 16; off > 0; off >>= 1) {
#pragma unroll
    for (int e = 0; e < E_NUM; ++e) acc[e] += __shfl_xor(acc[e], off, 32);
  }

  if (lane == 0) {
    int e0 = 0; float l0 = acc[0];
#pragma unroll
    for (int e = 1; e < E_NUM; ++e) if (acc[e] > l0) { l0 = acc[e]; e0 = e; }
    int e1 = (e0 == 0) ? 1 : 0; float l1 = acc[e1];
#pragma unroll
    for (int e = 0; e < E_NUM; ++e)
      if (e != e0 && acc[e] > l1) { l1 = acc[e]; e1 = e; }
    float p1  = __expf(l1 - l0);
    float inv = 1.0f / (1.0f + p1);

    int pos0 = atomicAdd(&cnt[e0], 1);
    toklist[e0 * T_TOK + pos0] = t;  wlist[e0 * T_TOK + pos0] = inv;
    int pos1 = atomicAdd(&cnt[e1], 1);
    toklist[e1 * T_TOK + pos1] = t;  wlist[e1 * T_TOK + pos1] = p1 * inv;
  }
}

// ---------------------------------------------------------------------------
// Fused expert FFN over pre-converted bf16 operands.
// ---------------------------------------------------------------------------
__global__ __launch_bounds__(256) void moe_ffn_kernel(
    const unsigned short* __restrict__ xbf,
    const unsigned short* __restrict__ W1T,   // [E][F][H] bf16
    const unsigned short* __restrict__ W2T,   // [E][H][F] bf16
    const float* __restrict__ b1, const float* __restrict__ b2,
    const int* __restrict__ cnt, const int* __restrict__ toklist,
    const float* __restrict__ wlist, float* __restrict__ out)
{
  __shared__ unsigned short Xs[TM * XLD];
  __shared__ unsigned short Hs[2][TM * HLD];   // double-buffered H chunk
  __shared__ int   toksS[TM];
  __shared__ float wtsS[TM];

  const int e     = blockIdx.y;
  const int tile0 = blockIdx.x * TM;
  const int n_e   = cnt[e];
  if (tile0 >= n_e) return;

  const int tid  = threadIdx.x;
  const int lane = tid & 31;
  const int wid  = tid >> 5;

  const unsigned short* W1Te = W1T + (size_t)e * F_DIM * H_DIM;
  const unsigned short* W2Te = W2T + (size_t)e * H_DIM * F_DIM;
  const float* b1e = b1 + (size_t)e * F_DIM;
  const float* b2e = b2 + (size_t)e * H_DIM;

  if (tid < TM) {
    int idx = tile0 + tid;
    int t = 0; float wt = 0.0f;
    if (idx < n_e) { t = toklist[e * T_TOK + idx]; wt = wlist[e * T_TOK + idx]; }
    toksS[tid] = t; wtsS[tid] = wt;
  }
  __syncthreads();

  // Gather 32 bf16 token rows into LDS: pure b128 copies.
  for (int i = tid; i < TM * (H_DIM / 8); i += 256) {
    int r = i / (H_DIM / 8);
    int q = i - r * (H_DIM / 8);
    *reinterpret_cast<uint4*>(&Xs[r * XLD + q * 8]) =
        *reinterpret_cast<const uint4*>(&xbf[(size_t)toksS[r] * H_DIM + q * 8]);
  }
  __syncthreads();

  const int mt1  = wid >> 2;        // GEMM1 subtile m-tile (0..1)
  const int nt1  = wid & 3;         // GEMM1 subtile n-tile (0..3)
  const int fn1  = nt1 * 16;        // GEMM1 column offset inside chunk
  const int n0w  = wid * 96;        // GEMM2: this wave's output columns

  v8f zero = {};
  v8f acc2[2][6];
#pragma unroll
  for (int a = 0; a < 2; ++a)
#pragma unroll
    for (int b = 0; b < 6; ++b) acc2[a][b] = zero;

  float b2v[6];
#pragma unroll
  for (int nt = 0; nt < 6; ++nt)
    b2v[nt] = b2e[n0w + nt * 16 + (lane & 15)];

  int buf = 0;
  for (int fc = 0; fc < F_DIM; fc += FN, buf ^= 1) {
    if (fc + FN < F_DIM) {   // hint next chunk's weights (global_prefetch_b8)
      __builtin_prefetch(W1Te + (size_t)(fc + FN + lane * 2) * H_DIM, 0, 1);
      __builtin_prefetch(W2Te + (size_t)(lane * 24) * F_DIM + fc + FN, 0, 1);
    }

    // ---- GEMM1: H-chunk(32x64) = X(32x768) * W1[:, fc:fc+64]
    // Fully unrolled with explicit next-step A/B double buffering.
    v8f h = zero;
    v16bf a_cur = frag_a_lds(Xs, XLD, mt1 * 16, 0, lane);
    v16bf b_cur = frag_b_glbT(W1Te, H_DIM, 0, fc + fn1, lane);
#pragma unroll
    for (int kc = 0; kc < H_DIM; kc += 32) {
      v16bf a_nxt = a_cur, b_nxt = b_cur;
      if (kc + 32 < H_DIM) {
        a_nxt = frag_a_lds(Xs, XLD, mt1 * 16, kc + 32, lane);
        b_nxt = frag_b_glbT(W1Te, H_DIM, kc + 32, fc + fn1, lane);
      }
      h = __builtin_amdgcn_wmma_f32_16x16x32_bf16(false, a_cur, false, b_cur,
                                                  (short)0, h, false, false);
      a_cur = a_nxt; b_cur = b_nxt;
    }
    float bias = b1e[fc + fn1 + (lane & 15)];

    // write H chunk into this chunk's buffer (safe: the other buffer is the
    // one still being read / was read last chunk; see barrier argument below)
    {
      unsigned short* hb = &Hs[buf][0];
      int col   = fn1 + (lane & 15);
      int rbase = mt1 * 16 + ((lane >> 4) << 3);
#pragma unroll
      for (int r = 0; r < 8; ++r)
        hb[(rbase + r) * HLD + col] = f2bf(gelu_tanh(h[r] + bias));
    }
    // ONE barrier per chunk: after it, every wave has (a) finished reading
    // buf^1 during the previous chunk's GEMM2 and (b) finished writing buf.
    __syncthreads();

    // ---- GEMM2: Y(32x768) += H-chunk(32x64) * W2[fc:fc+64, :]
    // Preload all four A fragments; rotate B one nt ahead.
    const unsigned short* hb = &Hs[buf][0];
    v16bf a00 = frag_a_lds(hb, HLD, 0,  0,  lane);
    v16bf a10 = frag_a_lds(hb, HLD, 16, 0,  lane);
    v16bf a01 = frag_a_lds(hb, HLD, 0,  32, lane);
    v16bf a11 = frag_a_lds(hb, HLD, 16, 32, lane);
    v16bf b_lo = frag_b_glbT(W2Te, F_DIM, fc, n0w, lane);
#pragma unroll
    for (int nt = 0; nt < 6; ++nt) {
      v16bf b0 = b_lo;
      v16bf b1f = frag_b_glbT(W2Te, F_DIM, fc + 32, n0w + nt * 16, lane);
      if (nt < 5)
        b_lo = frag_b_glbT(W2Te, F_DIM, fc, n0w + (nt + 1) * 16, lane);
      acc2[0][nt] = __builtin_amdgcn_wmma_f32_16x16x32_bf16(
          false, a00, false, b0, (short)0, acc2[0][nt], false, false);
      acc2[1][nt] = __builtin_amdgcn_wmma_f32_16x16x32_bf16(
          false, a10, false, b0, (short)0, acc2[1][nt], false, false);
      acc2[0][nt] = __builtin_amdgcn_wmma_f32_16x16x32_bf16(
          false, a01, false, b1f, (short)0, acc2[0][nt], false, false);
      acc2[1][nt] = __builtin_amdgcn_wmma_f32_16x16x32_bf16(
          false, a11, false, b1f, (short)0, acc2[1][nt], false, false);
    }
  }

  // ---- Weighted scatter: out[t] += w_t * (Y + b2)
#pragma unroll
  for (int mt = 0; mt < 2; ++mt) {
#pragma unroll
    for (int nt = 0; nt < 6; ++nt) {
#pragma unroll
      for (int r = 0; r < 8; ++r) {
        int m = mt * 16 + ((lane >> 4) << 3) + r;   // C/D: M = r + (L/16)*8
        float wt = wtsS[m];
        if (wt != 0.0f) {
          float v = (acc2[mt][nt][r] + b2v[nt]) * wt;
          unsafeAtomicAdd(out + (size_t)toksS[m] * H_DIM + n0w + nt * 16 + (lane & 15), v);
        }
      }
    }
  }
}

// ---------------------------------------------------------------------------
extern "C" void kernel_launch(void* const* d_in, const int* in_sizes, int n_in,
                              void* d_out, int out_size, void* d_ws, size_t ws_size,
                              hipStream_t stream) {
  const float* x      = (const float*)d_in[0];
  const float* gate_w = (const float*)d_in[1];
  const float* W1     = (const float*)d_in[2];
  const float* b1     = (const float*)d_in[3];
  const float* W2     = (const float*)d_in[4];
  const float* b2     = (const float*)d_in[5];
  float* out = (float*)d_out;

  // Workspace layout (all 64B aligned):
  //  [0)        16 x int   expert counters
  //  [64)       E*T int    token lists
  //  [+128K)    E*T float  combine weights
  //  [+128K)    T*H bf16   x in bf16
  //  [+6MB)     E*F*H bf16 W1 transposed  ([e][f][h])
  //  [+37.75MB) E*H*F bf16 W2 transposed  ([e][h][f])
  char* ws = (char*)d_ws;
  int*            cnt     = (int*)ws;
  int*            toklist = (int*)(ws + 64);
  float*          wlist   = (float*)(ws + 64 + (size_t)E_NUM * T_TOK * 4);
  unsigned short* xbf     = (unsigned short*)(ws + 64 + (size_t)E_NUM * T_TOK * 8);
  unsigned short* W1T     = xbf + (size_t)T_TOK * H_DIM;
  unsigned short* W2T     = W1T + (size_t)E_NUM * F_DIM * H_DIM;

  hipMemsetAsync(d_ws, 0, 16 * sizeof(int), stream);
  hipMemsetAsync(d_out, 0, (size_t)out_size * sizeof(float), stream);

  // one-time (per launch) precompute
  convert_x_kernel<<<(T_TOK * H_DIM / 2 + 255) / 256, 256, 0, stream>>>(
      x, (unsigned int*)xbf);
  transpose_to_bf16_kernel<<<dim3(F_DIM / 32, H_DIM / 32, E_NUM), 256, 0, stream>>>(
      W1, W1T, H_DIM, F_DIM);   // [H][F] -> [F][H]
  transpose_to_bf16_kernel<<<dim3(H_DIM / 32, F_DIM / 32, E_NUM), 256, 0, stream>>>(
      W2, W2T, F_DIM, H_DIM);   // [F][H] -> [H][F]

  moe_router_kernel<<<T_TOK / 8, 256, 0, stream>>>(x, gate_w, cnt, toklist, wlist);

  dim3 grid((T_TOK + TM - 1) / TM, E_NUM);
  moe_ffn_kernel<<<grid, 256, 0, stream>>>(xbf, W1T, W2T, b1, b2,
                                           cnt, toklist, wlist, out);
}
